// DMNETS_GNN_MB_54030688583948
// MI455X (gfx1250) — matrix operand order repaired and checked by reading, not compiled, not run on hardware
//
#include <hip/hip_runtime.h>
#include <math.h>

// ---------------------------------------------------------------------------
// DMNETS GNN mixture-of-Bernoulli loss, MI455X (gfx1250, wave32, WMMA).
// GEMMs: v_wmma_f32_16x16x32_f16, 64x16 output per wave, pre-packed B tiles.
// ---------------------------------------------------------------------------

typedef _Float16 f16;
typedef __attribute__((ext_vector_type(4)))  _Float16 v4h;
typedef __attribute__((ext_vector_type(8)))  _Float16 v8h;
typedef __attribute__((ext_vector_type(16))) _Float16 v16h;
typedef __attribute__((ext_vector_type(8)))  float    v8f;

#define NEG_SLOPE 0.2f

static inline unsigned cdiv_l(long a, long b){ return (unsigned)((a + b - 1) / b); }

// Monotonic float<->uint mapping so atomicMax(u32) == max(float).
__device__ __forceinline__ unsigned enc_ord(float f){
  unsigned u = __float_as_uint(f);
  return (u & 0x80000000u) ? ~u : (u | 0x80000000u);
}
__device__ __forceinline__ float dec_ord(unsigned u){
  return (u & 0x80000000u) ? __uint_as_float(u & 0x7fffffffu) : __uint_as_float(~u);
}

// ---------------- utility kernels ----------------
// n4 = n/4 (all tensors here have multiple-of-4 element counts)
__global__ void k_f32_to_f16_v4(const float* __restrict__ in, f16* __restrict__ out, long n4){
  long i = (long)blockIdx.x * blockDim.x + threadIdx.x;
  if (i < n4){
    const float4 x = ((const float4*)in)[i];
    v4h y; y[0] = (f16)x.x; y[1] = (f16)x.y; y[2] = (f16)x.z; y[3] = (f16)x.w;
    *(v4h*)(out + 4 * i) = y;
  }
}
__global__ void k_zero_f32(float* __restrict__ p, long n){
  long i = (long)blockIdx.x * blockDim.x + threadIdx.x;
  if (i < n) p[i] = 0.0f;
}
__global__ void k_elu(float* __restrict__ h, long n){
  long i = (long)blockIdx.x * blockDim.x + threadIdx.x;
  if (i < n){ float x = h[i]; h[i] = x > 0.0f ? x : __expf(x) - 1.0f; }
}

// ---------------- B-tile pre-pack ----------------
// Packed layout: tile (tn, kt) at dword offset (tn*(Kd/32)+kt)*256; within a
// tile, lane l owns dwords l*8..l*8+7; dword v = f16 pair (K=kb+2v, kb+2v+1)
// at column n0+l%16, with kb = kt*32 + 16*(l/16).  Columns >= Ncols are zero,
// so the GEMM inner loop needs no bounds checks.
__global__ void k_pack_w(const float* __restrict__ W, unsigned* __restrict__ Wp,
                         int Kd, int Ncols, int tilesN){
  int gid = blockIdx.x * blockDim.x + threadIdx.x;
  int ktiles = Kd >> 5;
  if (gid >= ktiles * tilesN * 256) return;
  int tile = gid >> 8, within = gid & 255;
  int lane = within >> 3, v = within & 7;
  int tn = tile / ktiles, kt = tile - tn * ktiles;
  int l15 = lane & 15, hi = lane >> 4;
  int k = kt * 32 + hi * 16 + v * 2;
  int n = tn * 16 + l15;
  unsigned short lo = 0, hu = 0;
  if (n < Ncols){
    lo = __builtin_bit_cast(unsigned short, (f16)W[(long)k * Ncols + n]);
    hu = __builtin_bit_cast(unsigned short, (f16)W[(long)(k + 1) * Ncols + n]);
  }
  Wp[gid] = (unsigned)lo | ((unsigned)hu << 16);
}

// ---------------- WMMA GEMM: C[M,Nc] = A[M,KD] @ W[KD,Nc] (+bias, relu) -----
// One wave computes a 64x16 output slab (4 stacked 16x16 WMMA tiles) so each
// B fragment is reused 4x. A is f16 row-major; B comes from the packed tiles.
// Fragment k order is contiguous: element s of lane half h holds k = 16h + s (A and B).
template<int KD>
__global__ __launch_bounds__(256) void k_gemm_f16(
    const f16* __restrict__ A, const unsigned* __restrict__ Wp,
    const float* __restrict__ bias, float* __restrict__ C,
    int Ncols, int tilesN, int totalWaves, int relu)
{
  int wid  = (int)((blockIdx.x * blockDim.x + threadIdx.x) >> 5);
  if (wid >= totalWaves) return;                 // wave-uniform exit
  int lane = (int)(threadIdx.x & 31);
  int tm = wid / tilesN, tn = wid - tm * tilesN;
  int m0 = tm << 6, n0 = tn << 4;
  int l15 = lane & 15, hi = lane >> 4;
  int ncol = n0 + l15;

  const unsigned* wp = Wp + ((long)tn * (KD / 32)) * 256 + lane * 8;
  const f16* abase = A + (long)(m0 + l15) * KD + hi * 16;

  v8f acc[4] = {};
#pragma unroll
  for (int kt = 0; kt < KD / 32; ++kt) {
    const int k0 = kt * 32;
    v16h b = *(const v16h*)(wp + kt * 256);      // 32B aligned, 2x b128
#pragma unroll
    for (int t = 0; t < 4; ++t) {
      const f16* ap = abase + (long)t * 16 * KD + k0;
      v8h alo = *(const v8h*)(ap);               // K = base..base+7
      v8h ahi = *(const v8h*)(ap + 8);           // K = base+8..base+15
      v16h a  = __builtin_shufflevector(alo, ahi,
                  0,1,2,3,4,5,6,7,8,9,10,11,12,13,14,15);
      acc[t] = __builtin_amdgcn_wmma_f32_16x16x32_f16(
          false, a, false, b, (short)0, acc[t], false, false);
    }
  }

  if (ncol < Ncols) {
    float bv = bias ? bias[ncol] : 0.0f;
#pragma unroll
    for (int t = 0; t < 4; ++t)
#pragma unroll
      for (int r = 0; r < 8; ++r) {
        float v = acc[t][r] + bv;
        if (relu) v = fmaxf(v, 0.0f);
        C[(long)(m0 + t * 16 + (hi << 3) + r) * Ncols + ncol] = v;
      }
  }
}

// ---------------- GAT attention kernels ----------------
__global__ void k_attn_self(const float* __restrict__ g,
                            const float* __restrict__ a_s, const float* __restrict__ a_d,
                            float* __restrict__ es, float* __restrict__ ed,
                            float* __restrict__ satt, unsigned* __restrict__ mxenc, int N){
  int gid = blockIdx.x * blockDim.x + threadIdx.x;
  if (gid >= N * 8) return;
  int n = gid >> 3, h = gid & 7;
  const float* gr = g + (long)n * 256 + h * 32;
  const float* as = a_s + h * 32;
  const float* ad = a_d + h * 32;
  float s = 0.0f, d = 0.0f;
#pragma unroll
  for (int f = 0; f < 32; ++f){ float x = gr[f]; s += x * as[f]; d += x * ad[f]; }
  es[gid] = s; ed[gid] = d;
  float v = s + d;
  v = v > 0.0f ? v : NEG_SLOPE * v;
  satt[gid]  = v;
  mxenc[gid] = enc_ord(v);
}

__global__ void k_edge_max(const int* __restrict__ src, const int* __restrict__ dst,
                           const float* __restrict__ es, const float* __restrict__ ed,
                           unsigned* __restrict__ mxenc, float* __restrict__ ebuf, int E){
  int gid = blockIdx.x * blockDim.x + threadIdx.x;
  if (gid >= E * 8) return;
  int e = gid >> 3, h = gid & 7;
  int s = src[e], d = dst[e];
  float v = es[s * 8 + h] + ed[d * 8 + h];
  v = v > 0.0f ? v : NEG_SLOPE * v;
  ebuf[gid] = v;
  atomicMax(&mxenc[d * 8 + h], enc_ord(v));
}

__global__ void k_edge_exp(const int* __restrict__ dst, const unsigned* __restrict__ mxenc,
                           float* __restrict__ ebuf, const float* __restrict__ satt,
                           float* __restrict__ selfex, float* __restrict__ den, int E, int N){
  int gid = blockIdx.x * blockDim.x + threadIdx.x;
  if (gid >= (E + N) * 8) return;
  int e = gid >> 3, h = gid & 7;
  if (e < E) {
    int d = dst[e];
    float ex = __expf(ebuf[gid] - dec_ord(mxenc[d * 8 + h]));
    ebuf[gid] = ex;
    atomicAdd(&den[d * 8 + h], ex);
  } else {
    int n = e - E;
    float ex = __expf(satt[n * 8 + h] - dec_ord(mxenc[n * 8 + h]));
    selfex[n * 8 + h] = ex;
    atomicAdd(&den[n * 8 + h], ex);
  }
}

__global__ void k_aggregate(const int* __restrict__ src, const int* __restrict__ dst,
                            const float* __restrict__ ebuf, const float* __restrict__ selfex,
                            const float* __restrict__ den, const float* __restrict__ g,
                            float* __restrict__ hout, int E){
  int b = blockIdx.x;
  int f = threadIdx.x;
  int h = f >> 5;
  int s, d; float num;
  if (b < E) { s = src[b]; d = dst[b]; num = ebuf[(long)b * 8 + h]; }
  else       { s = d = b - E;          num = selfex[(long)s * 8 + h]; }
  float w = num / den[(long)d * 8 + h];
  atomicAdd(&hout[(long)d * 256 + f], w * g[(long)s * 256 + f]);
}

// ---------------- gather / diffs ----------------
__global__ void k_gather_rows(const float* __restrict__ src, const int* __restrict__ idx,
                              float* __restrict__ out){
  int b = blockIdx.x, f = threadIdx.x;
  out[(long)b * 256 + f] = src[(long)idx[b] * 256 + f];
}
__global__ void k_diffs(const float* __restrict__ enc, const int* __restrict__ didx,
                        const float* __restrict__ dec, float* __restrict__ out){
  int b = blockIdx.x, f = threadIdx.x;
  out[(long)b * 256 + f] = enc[(long)didx[2 * b] * 256 + f] - dec[(long)b * 256 + f];
}

// ---------------- loss kernels ----------------
__global__ void k_mix_bce(const float* __restrict__ t1, const float* __restrict__ t2,
                          const float* __restrict__ al, const float* __restrict__ prev,
                          const float* __restrict__ lab, const int* __restrict__ sg,
                          float* __restrict__ red_adj, float* __restrict__ red_alpha, int M){
  int gid = blockIdx.x * blockDim.x + threadIdx.x;
  if (gid >= M * 20) return;
  int i = gid / 20, k = gid - i * 20;
  float p  = prev[i];
  float th = t1[gid] * p + t2[gid] * (1.0f - p);
  float y  = lab[i];
  float bce = fmaxf(th, 0.0f) - th * y + log1pf(__expf(-fabsf(th)));
  int s = sg[i];
  atomicAdd(&red_adj[s * 20 + k],   bce);
  atomicAdd(&red_alpha[s * 20 + k], al[gid]);
}

__global__ void k_subgraph(const float* __restrict__ red_adj, const float* __restrict__ red_alpha,
                           float* __restrict__ logp, int S){
  int s = blockIdx.x * blockDim.x + threadIdx.x;
  if (s >= S) return;
  float cnt = (float)((s % 127) + 1);   // subgraph j=b*127+(i-1) has i entries
  float ra[20], m1 = -1e30f;
#pragma unroll
  for (int k = 0; k < 20; ++k){ ra[k] = red_alpha[s * 20 + k] / cnt; m1 = fmaxf(m1, ra[k]); }
  float sum = 0.0f;
#pragma unroll
  for (int k = 0; k < 20; ++k) sum += __expf(ra[k] - m1);
  float lse = m1 + __logf(sum);
  float v[20], m2 = -1e30f;
#pragma unroll
  for (int k = 0; k < 20; ++k){ v[k] = (ra[k] - lse) - red_adj[s * 20 + k]; m2 = fmaxf(m2, v[k]); }
  sum = 0.0f;
#pragma unroll
  for (int k = 0; k < 20; ++k) sum += __expf(v[k] - m2);
  logp[s] = m2 + __logf(sum);
}

__global__ void k_final(const float* __restrict__ logp, float* __restrict__ out, int S){
  __shared__ float sm[256];
  float acc = 0.0f;
  for (int i = threadIdx.x; i < S; i += 256) acc += logp[i];
  sm[threadIdx.x] = acc;
  __syncthreads();
  for (int o = 128; o > 0; o >>= 1){
    if ((int)threadIdx.x < o) sm[threadIdx.x] += sm[threadIdx.x + o];
    __syncthreads();
  }
  if (threadIdx.x == 0) out[0] = -sm[0] / 16.0f;   // mean over B graphs
}

// ---------------- host-side orchestration ----------------
struct Scratch {
  f16 *a16;
  unsigned* wp;
  float *es, *ed, *satt, *selfex, *den, *ebuf;
  unsigned* mxenc;
};

static void launch_gemm(hipStream_t st, const f16* A16, const float* W, const float* bias,
                        float* C, int Mr, int Kd, int Nc, int relu, unsigned* wp){
  int tilesN = (Nc + 15) / 16;
  int packN  = (Kd / 32) * tilesN * 256;
  k_pack_w<<<cdiv_l(packN, 256), 256, 0, st>>>(W, wp, Kd, Nc, tilesN);
  int totalWaves = (Mr / 64) * tilesN;
  dim3 grid(cdiv_l(totalWaves, 8));              // 8 waves / block
  if (Kd == 128) k_gemm_f16<128><<<grid, 256, 0, st>>>(A16, wp, bias, C, Nc, tilesN, totalWaves, relu);
  else           k_gemm_f16<256><<<grid, 256, 0, st>>>(A16, wp, bias, C, Nc, tilesN, totalWaves, relu);
}

static void run_gat_layer(hipStream_t st, const float* h_in, int N, int din,
                          const float* W, const float* as, const float* ad,
                          const int* src, const int* dst, int E,
                          float* g, float* h_out, const Scratch& sc, int elu)
{
  long nA = (long)N * din;
  k_f32_to_f16_v4<<<cdiv_l(nA / 4, 256), 256, 0, st>>>(h_in, sc.a16, nA / 4);
  launch_gemm(st, sc.a16, W, nullptr, g, N, din, 256, 0, sc.wp);
  k_attn_self<<<cdiv_l((long)N * 8, 256), 256, 0, st>>>(g, as, ad, sc.es, sc.ed, sc.satt, sc.mxenc, N);
  k_edge_max <<<cdiv_l((long)E * 8, 256), 256, 0, st>>>(src, dst, sc.es, sc.ed, sc.mxenc, sc.ebuf, E);
  k_zero_f32 <<<cdiv_l((long)N * 8, 256), 256, 0, st>>>(sc.den, (long)N * 8);
  k_edge_exp <<<cdiv_l((long)(E + N) * 8, 256), 256, 0, st>>>(dst, sc.mxenc, sc.ebuf, sc.satt,
                                                              sc.selfex, sc.den, E, N);
  k_zero_f32 <<<cdiv_l((long)N * 256, 256), 256, 0, st>>>(h_out, (long)N * 256);
  k_aggregate<<<E + N, 256, 0, st>>>(src, dst, sc.ebuf, sc.selfex, sc.den, g, h_out, E);
  if (elu) k_elu<<<cdiv_l((long)N * 256, 256), 256, 0, st>>>(h_out, (long)N * 256);
}

extern "C" void kernel_launch(void* const* d_in, const int* in_sizes, int n_in,
                              void* d_out, int out_size, void* d_ws, size_t ws_size,
                              hipStream_t stream)
{
  (void)in_sizes; (void)n_in; (void)out_size; (void)ws_size;
  const int M = 130048, S_ = 2032, EX = 32768, EY = 262144, NM = 2048;

  const float* node_feat = (const float*)d_in[0];
  const float* labels    = (const float*)d_in[1];
  const float* prev      = (const float*)d_in[2];
  const float* encW[2]  = {(const float*)d_in[3],  (const float*)d_in[6]};
  const float* encAs[2] = {(const float*)d_in[4],  (const float*)d_in[7]};
  const float* encAd[2] = {(const float*)d_in[5],  (const float*)d_in[8]};
  const float* decW[2]  = {(const float*)d_in[9],  (const float*)d_in[12]};
  const float* decAs[2] = {(const float*)d_in[10], (const float*)d_in[13]};
  const float* decAd[2] = {(const float*)d_in[11], (const float*)d_in[14]};
  const float* mlpW[3][3]; const float* mlpB[3][3];   // theta1, theta2, alpha
  for (int m = 0; m < 3; ++m)
    for (int l = 0; l < 3; ++l){
      mlpW[m][l] = (const float*)d_in[15 + m * 6 + l * 2];
      mlpB[m][l] = (const float*)d_in[15 + m * 6 + l * 2 + 1];
    }
  const int* ex_src = (const int*)d_in[33]; const int* ex_dst = ex_src + EX;
  const int* ey_src = (const int*)d_in[34]; const int* ey_dst = ey_src + EY;
  const int* nfi    = (const int*)d_in[35];
  const int* didx   = (const int*)d_in[36];
  const int* sg     = (const int*)d_in[37];

  // ---- workspace carve (~530 MB) ----
  char* base = (char*)d_ws; size_t off = 0;
  auto carve = [&](size_t bytes) -> char* {
    char* p = base + off; off = (off + bytes + 255) & ~(size_t)255; return p;
  };
  float* big0 = (float*)carve((size_t)M * 256 * 4);
  float* big1 = (float*)carve((size_t)M * 256 * 4);
  float* big2 = (float*)carve((size_t)M * 256 * 4);
  Scratch sc;
  sc.a16    = (f16*)carve((size_t)M * 256 * 2);
  sc.wp     = (unsigned*)carve((size_t)8 * 16 * 256 * 4);   // max packed W
  sc.es     = (float*)carve((size_t)M * 8 * 4);
  sc.ed     = (float*)carve((size_t)M * 8 * 4);
  sc.satt   = (float*)carve((size_t)M * 8 * 4);
  sc.selfex = (float*)carve((size_t)M * 8 * 4);
  sc.den    = (float*)carve((size_t)M * 8 * 4);
  sc.mxenc  = (unsigned*)carve((size_t)M * 8 * 4);
  sc.ebuf   = (float*)carve((size_t)EY * 8 * 4);
  float* enc_out = (float*)carve((size_t)NM * 256 * 4);
  float* tout[3];
  for (int m = 0; m < 3; ++m) tout[m] = (float*)carve((size_t)M * 20 * 4);
  float* red_adj   = (float*)carve((size_t)S_ * 20 * 4);
  float* red_alpha = (float*)carve((size_t)S_ * 20 * 4);
  float* logp      = (float*)carve((size_t)S_ * 4);

  // ---- encoder GAT (N=2048) ----
  run_gat_layer(stream, node_feat, NM, 128, encW[0], encAs[0], encAd[0],
                ex_src, ex_dst, EX, big0, big1, sc, 1);
  run_gat_layer(stream, big1,      NM, 256, encW[1], encAs[1], encAd[1],
                ex_src, ex_dst, EX, big0, enc_out, sc, 0);

  // ---- gather enc rows, decoder GAT (N=M) ----
  k_gather_rows<<<M, 256, 0, stream>>>(enc_out, nfi, big0);
  run_gat_layer(stream, big0, M, 256, decW[0], decAs[0], decAd[0],
                ey_src, ey_dst, EY, big1, big2, sc, 1);
  run_gat_layer(stream, big2, M, 256, decW[1], decAs[1], decAd[1],
                ey_src, ey_dst, EY, big1, big0, sc, 0);   // dec result -> big0

  // ---- diffs = enc[diffs_idx] - dec  -> big2 ----
  k_diffs<<<M, 256, 0, stream>>>(enc_out, didx, big0, big2);

  // ---- three MLPs on diffs (theta1, theta2, alpha) ----
  const long n256 = (long)M * 256;
  for (int m = 0; m < 3; ++m) {
    k_f32_to_f16_v4<<<cdiv_l(n256 / 4, 256), 256, 0, stream>>>(big2, sc.a16, n256 / 4);
    launch_gemm(stream, sc.a16, mlpW[m][0], mlpB[m][0], big1, M, 256, 256, 1, sc.wp);
    k_f32_to_f16_v4<<<cdiv_l(n256 / 4, 256), 256, 0, stream>>>(big1, sc.a16, n256 / 4);
    launch_gemm(stream, sc.a16, mlpW[m][1], mlpB[m][1], big1, M, 256, 256, 1, sc.wp);
    k_f32_to_f16_v4<<<cdiv_l(n256 / 4, 256), 256, 0, stream>>>(big1, sc.a16, n256 / 4);
    launch_gemm(stream, sc.a16, mlpW[m][2], mlpB[m][2], tout[m], M, 256, 20, 0, sc.wp);
  }

  // ---- loss: BCE mixture + sorted segment sums + log-softmax/LSE + mean ----
  k_zero_f32<<<cdiv_l((long)S_ * 20, 256), 256, 0, stream>>>(red_adj,   (long)S_ * 20);
  k_zero_f32<<<cdiv_l((long)S_ * 20, 256), 256, 0, stream>>>(red_alpha, (long)S_ * 20);
  k_mix_bce<<<cdiv_l((long)M * 20, 256), 256, 0, stream>>>(tout[0], tout[1], tout[2],
                                                           prev, labels, sg, red_adj, red_alpha, M);
  k_subgraph<<<cdiv_l(S_, 256), 256, 0, stream>>>(red_adj, red_alpha, logp, S_);
  k_final<<<1, 256, 0, stream>>>(logp, (float*)d_out, S_);
}
